// CrossAttention_5978594476652
// MI455X (gfx1250) — compile-verified, matrix operand order checked by reading, not run on hardware
//
#include <hip/hip_runtime.h>

#define DIMX   768
#define NHEADS 8
#define HD     96
#define SEQ    1024
#define NB     8
#define MTOT   (NB * SEQ)   // 8192

typedef __attribute__((ext_vector_type(16))) __bf16 v16bf;
typedef __attribute__((ext_vector_type(8)))  float  v8f;

union Frag {
    v16bf v;
    uint4 q[2];
    unsigned short s[16];
};
union AccU {
    v8f   v;
    float f[8];
};
union BfBits {
    __bf16 b;
    unsigned short u;
};

// ---------------------------------------------------------------------------
// fp32 -> bf16 convert (vector by 4)
// ---------------------------------------------------------------------------
__global__ void cvt_bf16_kernel(const float* __restrict__ x,
                                __bf16* __restrict__ y, int n) {
    int i = (blockIdx.x * blockDim.x + threadIdx.x) * 4;
    if (i < n) {
        float4 f = *(const float4*)(x + i);
        union { __bf16 b[4]; uint2 u; } o;
        o.b[0] = (__bf16)f.x; o.b[1] = (__bf16)f.y;
        o.b[2] = (__bf16)f.z; o.b[3] = (__bf16)f.w;
        *(uint2*)(y + i) = o.u;
    }
}

// ---------------------------------------------------------------------------
// bf16 WMMA GEMM:  C[m,n] = sum_k A[m,k] * W[n,k]   (i.e. y = x @ W^T)
// A: [MTOT x DIMX] bf16 row-major,  W: [DIMX x DIMX] bf16 row-major.
// Block = 256 thr (8 waves) -> 128x128 tile; wave -> 32x64 (2x4 WMMA tiles).
// MODE 0: store bf16 [M x DIMX]
// MODE 1: store bf16 transposed per head: Vt[n,h,d,s]
// MODE 2: store f32 + bias -> d_out
// ---------------------------------------------------------------------------
template <int MODE>
__global__ __launch_bounds__(256)
void gemm_bf16_kernel(const __bf16* __restrict__ A,
                      const __bf16* __restrict__ W,
                      void* __restrict__ out,
                      const float* __restrict__ bias) {
    const int lane  = threadIdx.x & 31;
    const int wid   = threadIdx.x >> 5;
    const int wr    = wid & 3;          // 4 row-waves
    const int wc    = wid >> 2;         // 2 col-waves
    const int rowBase = blockIdx.y * 128 + wr * 32;
    const int colBase = blockIdx.x * 128 + wc * 64;
    const int l15   = lane & 15;
    const int lhalf = lane >> 4;
    const int kb    = lhalf * 8;        // per-half K sub-offset (A/B layout)

    AccU acc[2][4];
#pragma unroll
    for (int i = 0; i < 2; i++)
#pragma unroll
        for (int t = 0; t < 4; t++)
#pragma unroll
            for (int e = 0; e < 8; e++) acc[i][t].f[e] = 0.0f;

    for (int k = 0; k < DIMX; k += 32) {
        Frag a[2], b[4];
#pragma unroll
        for (int i = 0; i < 2; i++) {
            const __bf16* p = A + (size_t)(rowBase + i * 16 + l15) * DIMX + k + kb;
            a[i].q[0] = *(const uint4*)p;
            a[i].q[1] = *(const uint4*)(p + 16);
        }
#pragma unroll
        for (int t = 0; t < 4; t++) {
            const __bf16* p = W + (size_t)(colBase + t * 16 + l15) * DIMX + k + kb;
            b[t].q[0] = *(const uint4*)p;
            b[t].q[1] = *(const uint4*)(p + 16);
        }
#pragma unroll
        for (int i = 0; i < 2; i++)
#pragma unroll
            for (int t = 0; t < 4; t++)
                acc[i][t].v = __builtin_amdgcn_wmma_f32_16x16x32_bf16(
                    false, a[i].v, false, b[t].v, (short)0, acc[i][t].v,
                    false, false);
    }

    // C layout: VGPR j -> row j (lanes 0-15) / row j+8 (lanes 16-31), col = l15
#pragma unroll
    for (int i = 0; i < 2; i++)
#pragma unroll
        for (int t = 0; t < 4; t++)
#pragma unroll
            for (int j = 0; j < 8; j++) {
                int row = rowBase + i * 16 + j + lhalf * 8;
                int col = colBase + t * 16 + l15;
                float v = acc[i][t].f[j];
                if (MODE == 0) {
                    ((__bf16*)out)[(size_t)row * DIMX + col] = (__bf16)v;
                } else if (MODE == 1) {
                    int nb = row >> 10, sr = row & 1023;
                    int h = col / HD, d = col % HD;
                    ((__bf16*)out)[(((size_t)nb * NHEADS + h) * HD + d) * SEQ + sr] =
                        (__bf16)v;
                } else {
                    ((float*)out)[(size_t)row * DIMX + col] = v + bias[col];
                }
            }
}

// ---------------------------------------------------------------------------
// Flash-style attention.  Block = 256 thr (8 waves) handles one (n,h,qblk of
// 128 rows); each wave owns 16 q-rows.  Online softmax, bf16 WMMA throughout.
// Q,K: [N,S,D] bf16 ; Vt: [N,H,hd,S] bf16 ; O: [N,S,D] bf16 (heads concat)
// ---------------------------------------------------------------------------
__global__ __launch_bounds__(256)
void attn_kernel(const __bf16* __restrict__ Q, const __bf16* __restrict__ K,
                 const __bf16* __restrict__ Vt, __bf16* __restrict__ O) {
    __shared__ __align__(16) unsigned short plds[8 * 16 * 32];

    const int lane  = threadIdx.x & 31;
    const int wid   = threadIdx.x >> 5;
    const int l15   = lane & 15;
    const int lhalf = lane >> 4;
    const int kb8   = lhalf * 8;

    const int blk  = blockIdx.x;
    const int qblk = blk & 7;           // S/128 = 8
    const int h    = (blk >> 3) & 7;
    const int n    = blk >> 6;
    const int qbase = qblk * 128 + wid * 16;
    unsigned short* myLds = plds + wid * (16 * 32);

    // Preload this wave's Q fragments (3 chunks of K=32 covering hd=96)
    Frag qf[3];
#pragma unroll
    for (int c = 0; c < 3; c++) {
        const __bf16* p =
            Q + (size_t)(n * SEQ + qbase + l15) * DIMX + h * HD + c * 32 + kb8;
        qf[c].q[0] = *(const uint4*)p;
        qf[c].q[1] = *(const uint4*)(p + 16);
    }

    AccU o[6];
#pragma unroll
    for (int t = 0; t < 6; t++)
#pragma unroll
        for (int e = 0; e < 8; e++) o[t].f[e] = 0.0f;
    float m[8], l[8];
#pragma unroll
    for (int j = 0; j < 8; j++) { m[j] = -1e30f; l[j] = 0.0f; }

    const float scale = 0.0360843918243516f;   // 1/sqrt(768)

    for (int kk = 0; kk < SEQ; kk += 32) {
        AccU s0, s1;
#pragma unroll
        for (int e = 0; e < 8; e++) { s0.f[e] = 0.0f; s1.f[e] = 0.0f; }

        // scores: Q[16x96] x K^T chunks -> two 16x16 tiles (cols kk..+15, +16..+31)
#pragma unroll
        for (int c = 0; c < 3; c++) {
            Frag kf0, kf1;
            const __bf16* p0 =
                K + (size_t)(n * SEQ + kk + l15) * DIMX + h * HD + c * 32 + kb8;
            kf0.q[0] = *(const uint4*)p0;
            kf0.q[1] = *(const uint4*)(p0 + 16);
            const __bf16* p1 = p0 + (size_t)16 * DIMX;
            kf1.q[0] = *(const uint4*)p1;
            kf1.q[1] = *(const uint4*)(p1 + 16);
            s0.v = __builtin_amdgcn_wmma_f32_16x16x32_bf16(
                false, qf[c].v, false, kf0.v, (short)0, s0.v, false, false);
            s1.v = __builtin_amdgcn_wmma_f32_16x16x32_bf16(
                false, qf[c].v, false, kf1.v, (short)0, s1.v, false, false);
        }

        // online softmax; row j lives in lanes<16 (row j) / lanes>=16 (row j+8)
#pragma unroll
        for (int j = 0; j < 8; j++) {
            float a0 = s0.f[j] * scale;
            float a1 = s1.f[j] * scale;
            float mx = fmaxf(a0, a1);
            mx = fmaxf(mx, __shfl_xor(mx, 1, 32));
            mx = fmaxf(mx, __shfl_xor(mx, 2, 32));
            mx = fmaxf(mx, __shfl_xor(mx, 4, 32));
            mx = fmaxf(mx, __shfl_xor(mx, 8, 32));
            float mn = fmaxf(m[j], mx);
            float fr = __expf(m[j] - mn);
            m[j] = mn;
            l[j] *= fr;
#pragma unroll
            for (int t = 0; t < 6; t++) o[t].f[j] *= fr;
            float p0 = __expf(a0 - mn);
            float p1 = __expf(a1 - mn);
            l[j] += p0 + p1;   // per-lane partial row sum (reduced at the end)
            // repack P tile -> LDS row-major [16 x 32] (wave-private)
            BfBits c0; c0.b = (__bf16)p0;
            BfBits c1; c1.b = (__bf16)p1;
            int row = j + 8 * lhalf;
            myLds[row * 32 + l15]      = c0.u;
            myLds[row * 32 + 16 + l15] = c1.u;
        }

        // read P back in A-fragment layout (same-wave LDS, in-order, no barrier)
        Frag pa;
        const unsigned short* lp = myLds + l15 * 32 + kb8;
        pa.q[0] = *(const uint4*)lp;
        pa.q[1] = *(const uint4*)(lp + 16);

        // out += P[16x32] x V[32x96]  (Vt rows are contiguous in s)
#pragma unroll
        for (int t = 0; t < 6; t++) {
            Frag vf;
            const __bf16* p =
                Vt + ((size_t)(n * NHEADS + h) * HD + t * 16 + l15) * SEQ + kk + kb8;
            vf.q[0] = *(const uint4*)p;
            vf.q[1] = *(const uint4*)(p + 16);
            o[t].v = __builtin_amdgcn_wmma_f32_16x16x32_bf16(
                false, pa.v, false, vf.v, (short)0, o[t].v, false, false);
        }
    }

    // normalize and store (heads concatenated back into [N,S,D])
#pragma unroll
    for (int j = 0; j < 8; j++) {
        float ls = l[j];
        ls += __shfl_xor(ls, 1, 32);
        ls += __shfl_xor(ls, 2, 32);
        ls += __shfl_xor(ls, 4, 32);
        ls += __shfl_xor(ls, 8, 32);
        float inv = 1.0f / ls;
        int row = qbase + j + 8 * lhalf;
#pragma unroll
        for (int t = 0; t < 6; t++) {
            float v = o[t].f[j] * inv;
            O[(size_t)(n * SEQ + row) * DIMX + h * HD + t * 16 + l15] = (__bf16)v;
        }
    }
}

// ---------------------------------------------------------------------------
extern "C" void kernel_launch(void* const* d_in, const int* in_sizes, int n_in,
                              void* d_out, int out_size, void* d_ws,
                              size_t ws_size, hipStream_t stream) {
    (void)in_sizes; (void)n_in; (void)out_size; (void)ws_size;
    const float* q  = (const float*)d_in[0];
    const float* kv = (const float*)d_in[1];
    const float* wq = (const float*)d_in[2];
    const float* wk = (const float*)d_in[3];
    const float* wv = (const float*)d_in[4];
    const float* wp = (const float*)d_in[5];
    const float* bp = (const float*)d_in[6];

    char* ws = (char*)d_ws;
    size_t off = 0;
    const size_t ACT = (size_t)MTOT * DIMX * 2;   // 12.58 MB
    const size_t WB  = (size_t)DIMX * DIMX * 2;   // 1.18 MB
    __bf16* qb  = (__bf16*)(ws + off); off += ACT;
    __bf16* kvb = (__bf16*)(ws + off); off += ACT;
    __bf16* wqb = (__bf16*)(ws + off); off += WB;
    __bf16* wkb = (__bf16*)(ws + off); off += WB;
    __bf16* wvb = (__bf16*)(ws + off); off += WB;
    __bf16* wpb = (__bf16*)(ws + off); off += WB;
    __bf16* Qb  = (__bf16*)(ws + off); off += ACT;
    __bf16* Kb  = (__bf16*)(ws + off); off += ACT;
    __bf16* Vtb = (__bf16*)(ws + off); off += ACT;
    __bf16* Ab  = (__bf16*)(ws + off); off += ACT;

    const int nact = MTOT * DIMX;   // 6291456 (divisible by 1024)
    const int nw   = DIMX * DIMX;   // 589824  (divisible by 1024)
    cvt_bf16_kernel<<<nact / 4 / 256, 256, 0, stream>>>(q, qb, nact);
    cvt_bf16_kernel<<<nact / 4 / 256, 256, 0, stream>>>(kv, kvb, nact);
    cvt_bf16_kernel<<<nw / 4 / 256, 256, 0, stream>>>(wq, wqb, nw);
    cvt_bf16_kernel<<<nw / 4 / 256, 256, 0, stream>>>(wk, wkb, nw);
    cvt_bf16_kernel<<<nw / 4 / 256, 256, 0, stream>>>(wv, wvb, nw);
    cvt_bf16_kernel<<<nw / 4 / 256, 256, 0, stream>>>(wp, wpb, nw);

    dim3 gg(DIMX / 128, MTOT / 128);   // (6, 64)
    gemm_bf16_kernel<0><<<gg, 256, 0, stream>>>(qb,  wqb, Qb,  nullptr);
    gemm_bf16_kernel<0><<<gg, 256, 0, stream>>>(kvb, wkb, Kb,  nullptr);
    gemm_bf16_kernel<1><<<gg, 256, 0, stream>>>(kvb, wvb, Vtb, nullptr);

    attn_kernel<<<NB * NHEADS * (SEQ / 128), 256, 0, stream>>>(Qb, Kb, Vtb, Ab);

    gemm_bf16_kernel<2><<<gg, 256, 0, stream>>>(Ab, wpb, d_out, bp);
}
